// ChannelAffinityAttention_33423435498335
// MI455X (gfx1250) — matrix operand order repaired and checked by reading, not compiled, not run on hardware
//
#include <hip/hip_runtime.h>
#include <hip/hip_bf16.h>

typedef __attribute__((ext_vector_type(16))) __bf16 v16bf;
typedef __attribute__((ext_vector_type(8)))  float  v8f;

#define BATCH 16
#define CCH   256
#define HW    16384   // 128*128
#define NTOK  64
#define DDIM  768
#define MIDD  64

// ---------------- Kernel A: global average pool over H,W -> vis[b,c] ----------------
__global__ __launch_bounds__(256)
void caa_pool_kernel(const float* __restrict__ vf, float* __restrict__ vis) {
    const int bc = blockIdx.x;                 // b*C + c
    const float4* p = reinterpret_cast<const float4*>(vf + (size_t)bc * HW);
    float s = 0.f;
    for (int i = threadIdx.x; i < HW / 4; i += 256) {
        float4 q = p[i];
        s += q.x + q.y + q.z + q.w;
    }
    __shared__ float red[256];
    red[threadIdx.x] = s;
    __syncthreads();
    for (int off = 128; off > 0; off >>= 1) {
        if ((int)threadIdx.x < off) red[threadIdx.x] += red[threadIdx.x + off];
        __syncthreads();
    }
    if (threadIdx.x == 0) vis[bc] = red[0] * (1.f / (float)HW);
}

// ------------- Kernel B: text projection (WMMA) + per-channel softmax + sigmoid -------------
__global__ __launch_bounds__(128)
void caa_proj_attn_kernel(const float* __restrict__ text_feat,   // [B,N,D]
                          const int*   __restrict__ text_mask,   // [B,N]
                          const float* __restrict__ W1,          // [D,MID]
                          const float* __restrict__ b1,          // [MID]
                          const float* __restrict__ W2,          // [MID,C]
                          const float* __restrict__ b2,          // [C]
                          const float* __restrict__ vis,         // [B,C]
                          const float* __restrict__ temperature, // [1]
                          float*       __restrict__ sig)         // [B,C]
{
    const int b    = blockIdx.x;
    const int tid  = threadIdx.x;
    const int wave = tid >> 5;        // 4 waves
    const int lane = tid & 31;
    const int g    = lane >> 4;       // lane group (0/1)
    const int hl   = lane & 15;

    __shared__ __bf16 hidden[NTOK][MIDD + 8];     // relu(text @ W1 + b1), [token][mid]
    __shared__ __bf16 tprojT[CCH][NTOK + 2];      // text_proj transposed: [channel][token]

    // ---------------- GEMM1: hidden = relu(text_feat[b] @ W1 + b1), 64x64 ----------------
    // wave w computes token rows [16w, 16w+16), all 64 mid columns (4 N-tiles)
    {
        v8f acc[4] = {};
        const float* arow = text_feat + ((size_t)b * NTOK + wave * 16 + hl) * DDIM;
        for (int k0 = 0; k0 < DDIM; k0 += 32) {
            // A fragment: 16-bit A 16x32 layout
            v16bf a;
#pragma unroll
            for (int v = 0; v < 8; ++v)
#pragma unroll
                for (int h = 0; h < 2; ++h) {
                    int k = ((v >= 4) ? 16 : 0) + ((v & 3) * 2 + h) + g * 8;
                    a[v * 2 + h] = (__bf16)arow[k0 + k];
                }
#pragma unroll
            for (int nt = 0; nt < 4; ++nt) {
                // B fragment: 16-bit B 32x16 layout
                v16bf bb;
#pragma unroll
                for (int v = 0; v < 8; ++v)
#pragma unroll
                    for (int h = 0; h < 2; ++h) {
                        int k = ((v >= 4) ? 16 : 0) + ((v & 3) * 2 + h) + g * 8;
                        bb[v * 2 + h] = (__bf16)W1[(size_t)(k0 + k) * MIDD + nt * 16 + hl];
                    }
                acc[nt] = __builtin_amdgcn_wmma_f32_16x16x32_bf16(
                    false, a, false, bb, (short)0, acc[nt], false, false);
            }
        }
        // bias + ReLU, store bf16 hidden
#pragma unroll
        for (int nt = 0; nt < 4; ++nt)
#pragma unroll
            for (int r = 0; r < 8; ++r) {
                int m = wave * 16 + g * 8 + r;   // token
                int n = nt * 16 + hl;            // mid channel
                float vv = acc[nt][r] + b1[n];
                hidden[m][n] = (__bf16)fmaxf(vv, 0.f);
            }
    }
    __syncthreads();

    // ---------------- GEMM2: text_proj = hidden @ W2 + b2, 64x256 (stored transposed) ----------------
    {
        // A fragments for K=0..31 and K=32..63 (rows = tokens of this wave's M-tile)
        v16bf a2[2];
#pragma unroll
        for (int kk = 0; kk < 2; ++kk)
#pragma unroll
            for (int v = 0; v < 8; ++v)
#pragma unroll
                for (int h = 0; h < 2; ++h) {
                    int k = kk * 32 + ((v >= 4) ? 16 : 0) + ((v & 3) * 2 + h) + g * 8;
                    a2[kk][v * 2 + h] = hidden[wave * 16 + hl][k];
                }
#pragma unroll 4
        for (int nt = 0; nt < 16; ++nt) {
            v8f acc2 = {};
#pragma unroll
            for (int kk = 0; kk < 2; ++kk) {
                v16bf bb;
#pragma unroll
                for (int v = 0; v < 8; ++v)
#pragma unroll
                    for (int h = 0; h < 2; ++h) {
                        int k = kk * 32 + ((v >= 4) ? 16 : 0) + ((v & 3) * 2 + h) + g * 8;
                        bb[v * 2 + h] = (__bf16)W2[(size_t)k * CCH + nt * 16 + hl];
                    }
                acc2 = __builtin_amdgcn_wmma_f32_16x16x32_bf16(
                    false, a2[kk], false, bb, (short)0, acc2, false, false);
            }
            int c = nt * 16 + hl;
            float bias = b2[c];
#pragma unroll
            for (int r = 0; r < 8; ++r) {
                int tok = wave * 16 + g * 8 + r;
                tprojT[c][tok] = (__bf16)(acc2[r] + bias);
            }
        }
    }
    __syncthreads();

    // ---------------- Phase 3: masked softmax over tokens + weighted sum + sigmoid ----------------
    const float temp  = temperature[0];
    const int*  maskb = text_mask + b * NTOK;
    for (int c = tid; c < CCH; c += 128) {
        float scale = vis[b * CCH + c] / temp;
        float mx = -__builtin_inff();
#pragma unroll 8
        for (int n = 0; n < NTOK; ++n) {
            float tp  = (float)tprojT[c][n];
            float aff = (maskb[n] != 0) ? scale * tp : -__builtin_inff();
            mx = fmaxf(mx, aff);
        }
        float se = 0.f, sw = 0.f;
#pragma unroll 8
        for (int n = 0; n < NTOK; ++n) {
            float tp = (float)tprojT[c][n];
            if (maskb[n] != 0) {
                float e = __expf(scale * tp - mx);
                se += e;
                sw += e * tp;
            }
        }
        float cw = sw / se;
        sig[b * CCH + c] = 1.f / (1.f + __expf(-cw));
    }
}

// ---------------- Kernel C: out = visual_feat * sigmoid(cw)[b,c] ----------------
__global__ __launch_bounds__(256)
void caa_scale_kernel(const float* __restrict__ vf,
                      const float* __restrict__ sig,
                      float*       __restrict__ out) {
    size_t i  = (size_t)blockIdx.x * 256 + threadIdx.x;  // float4 index
    int    bc = (int)(i >> 12);                          // 4096 float4 per (b,c) plane
    float  s  = sig[bc];
    float4 q  = reinterpret_cast<const float4*>(vf)[i];
    q.x *= s; q.y *= s; q.z *= s; q.w *= s;
    reinterpret_cast<float4*>(out)[i] = q;
}

extern "C" void kernel_launch(void* const* d_in, const int* in_sizes, int n_in,
                              void* d_out, int out_size, void* d_ws, size_t ws_size,
                              hipStream_t stream) {
    const float* visual_feat = (const float*)d_in[0];  // [B,C,H,W]
    const float* text_feat   = (const float*)d_in[1];  // [B,N,D]
    const int*   text_mask   = (const int*)  d_in[2];  // [B,N]
    const float* W1          = (const float*)d_in[3];  // [D,MID]
    const float* b1          = (const float*)d_in[4];  // [MID]
    const float* W2          = (const float*)d_in[5];  // [MID,C]
    const float* b2          = (const float*)d_in[6];  // [C]
    const float* temperature = (const float*)d_in[7];  // [1]
    float*       out         = (float*)d_out;

    float* vis = (float*)d_ws;              // [B*C]
    float* sg  = vis + BATCH * CCH;         // [B*C]

    // 1) global average pool
    caa_pool_kernel<<<BATCH * CCH, 256, 0, stream>>>(visual_feat, vis);

    // 2) text projection (WMMA) + channel attention weights
    caa_proj_attn_kernel<<<BATCH, 128, 0, stream>>>(text_feat, text_mask, W1, b1,
                                                    W2, b2, vis, temperature, sg);

    // 3) scale visual features
    const size_t nvec4 = (size_t)BATCH * CCH * HW / 4;
    caa_scale_kernel<<<(unsigned)(nvec4 / 256), 256, 0, stream>>>(visual_feat, sg, out);
}
